// MultiHeadAttention_15822659518597
// MI455X (gfx1250) — matrix operand order repaired and checked by reading, not compiled, not run on hardware
//
#include <hip/hip_runtime.h>

// CDNA5 / gfx1250, wave32. Split-precision f16 WMMA GEMM pipeline, LDS-tiled.
//
// Refactored computation (v and softmax are dead in the reference):
//   Stage 0: Wrep[n][j][d] = fc_w[d][j*16+n]                 (64 MiB transpose, optional)
//   Stage A: W2[b][n*64+kk][d] = sum_j K[b,j,n,kk] * fc_w[d, j*16+n]   (16 MiB in d_ws)
//   Stage B: out[b][i][d]      = 0.125 * sum_c q[b,i,c] * W2[b][c][d] + fc_b[d]
//
// fp32 accuracy via x = hi(f16) + lo(f16): C += Ah*Bh + Ah*Bl + Al*Bh (f32 accum).

typedef __attribute__((ext_vector_type(16))) _Float16 v16h;
typedef __attribute__((ext_vector_type(8)))  _Float16 v8h;
typedef __attribute__((ext_vector_type(8)))  float    v8f;

#define DMODEL 1024
#define NHEADS 16
#define DKH    64
#define BATCH  4
#define SEQ    1024   // LQ == LK

// ---------------------------------------------------------------------------
// Stage 0: LDS-tiled transpose  fc_w[d][x] -> Wrep[n][j][d], x = j*16+n.
// ---------------------------------------------------------------------------
__global__ __launch_bounds__(256)
void repack_fcw(const float* __restrict__ W, float* __restrict__ Wrep) {
    __shared__ float tile[32][33];
    const int tx = threadIdx.x;          // 0..31
    const int ty = threadIdx.y;          // 0..7
    const int x0 = blockIdx.x * 32;      // over 16384
    const int d0 = blockIdx.y * 32;      // over 1024
    #pragma unroll
    for (int rr = 0; rr < 4; ++rr) {
        const int dl = ty + rr * 8;
        tile[dl][tx] = W[(size_t)(d0 + dl) * (SEQ * NHEADS) + (x0 + tx)];
    }
    __syncthreads();
    #pragma unroll
    for (int rr = 0; rr < 4; ++rr) {
        const int xl = ty + rr * 8;
        const int x  = x0 + xl;
        const int n  = x & 15, j = x >> 4;
        Wrep[((size_t)n * SEQ + j) * DMODEL + (d0 + tx)] = tile[tx][xl];
    }
}

// ---------------------------------------------------------------------------
// Shared 64x64 macro-tile GEMM core. blockDim.x == 128 (4 waves).
//   C[m,n] (+)= sum_k A[m,k]*B[k,n],  A[m,k]=A0[m*sAm + k*sAk],
//                                     B[k,n]=B0[k*sBk + n*sBn]
// Wave w accumulates rows 16w..16w+15 into acc[0..3] (cols 0..63).
// LDS holds split-f16 tiles pre-permuted into the WMMA fragment layout.
// ---------------------------------------------------------------------------
__device__ __forceinline__ void gemm_macro64(
    const float* __restrict__ A0, long sAm, long sAk,
    const float* __restrict__ B0, long sBk, long sBn,
    int kTiles, v8f acc[4])
{
    __shared__ __align__(32) _Float16 Ah[2][4 * 32 * 16];
    __shared__ __align__(32) _Float16 Al[2][4 * 32 * 16];
    __shared__ __align__(32) _Float16 Bh[2][4 * 32 * 16];
    __shared__ __align__(32) _Float16 Bl[2][4 * 32 * 16];

    const int tid  = threadIdx.x;
    const int lane = tid & 31;
    const int wave = tid >> 5;

    // Producer roles (128 threads):
    const int am  = tid & 63;   // A tile row 0..63
    const int au  = tid >> 6;   // A k-half (k = 16*au + i)
    const int bn  = tid & 63;   // B tile col 0..63
    const int bkh = tid >> 6;   // B k-half

    const v8f zero = {};
    acc[0] = zero; acc[1] = zero; acc[2] = zero; acc[3] = zero;

    for (int kt = 0; kt < kTiles; ++kt) {
        const int  buf = kt & 1;
        const long k0  = (long)kt * 32;

        // ---- A fill: row am, k = k0 + 16*au + i, i = 0..15 ----------------
        {
            float x[16];
            #pragma unroll
            for (int i = 0; i < 16; ++i)
                x[i] = A0[(long)am * sAm + (k0 + 16 * au + i) * sAk];
            v8h h0, h1, l0, l1;     // i<8 -> lane-half 0 ; i>=8 -> lane-half 1
            #pragma unroll
            for (int i = 0; i < 8; ++i) {
                _Float16 hi = (_Float16)x[i];
                h0[i] = hi; l0[i] = (_Float16)(x[i] - (float)hi);
                hi = (_Float16)x[8 + i];
                h1[i] = hi; l1[i] = (_Float16)(x[8 + i] - (float)hi);
            }
            const int mt = am >> 4, r = am & 15;
            const int base0 = (mt * 32 + r)      * 16 + 8 * au;  // h = 0
            const int base1 = (mt * 32 + r + 16) * 16 + 8 * au;  // h = 1
            *(v8h*)&Ah[buf][base0] = h0;  *(v8h*)&Al[buf][base0] = l0;
            *(v8h*)&Ah[buf][base1] = h1;  *(v8h*)&Al[buf][base1] = l1;
        }
        // ---- B fill: col bn, k = k0 + 16*bkh + i, i = 0..15 ---------------
        {
            float x[16];
            #pragma unroll
            for (int i = 0; i < 16; ++i)
                x[i] = B0[(k0 + 16 * bkh + i) * sBk + (long)bn * sBn];
            v8h h0, h1, l0, l1;
            #pragma unroll
            for (int i = 0; i < 8; ++i) {
                _Float16 hi = (_Float16)x[i];
                h0[i] = hi; l0[i] = (_Float16)(x[i] - (float)hi);
                hi = (_Float16)x[8 + i];
                h1[i] = hi; l1[i] = (_Float16)(x[8 + i] - (float)hi);
            }
            const int nt = bn >> 4, c = bn & 15;
            const int base0 = (nt * 32 + c)      * 16 + 8 * bkh;  // h = 0
            const int base1 = (nt * 32 + c + 16) * 16 + 8 * bkh;  // h = 1
            *(v8h*)&Bh[buf][base0] = h0;  *(v8h*)&Bl[buf][base0] = l0;
            *(v8h*)&Bh[buf][base1] = h1;  *(v8h*)&Bl[buf][base1] = l1;
        }

        __syncthreads();   // fill(kt) visible; buf kt&1 was last read at kt-2

        // ---- consume: fragment-ready 32B/lane loads -----------------------
        const v16h a_h = *(const v16h*)&Ah[buf][(wave * 32 + lane) * 16];
        const v16h a_l = *(const v16h*)&Al[buf][(wave * 32 + lane) * 16];
        #pragma unroll
        for (int nt = 0; nt < 4; ++nt) {
            const v16h b_h = *(const v16h*)&Bh[buf][(nt * 32 + lane) * 16];
            const v16h b_l = *(const v16h*)&Bl[buf][(nt * 32 + lane) * 16];
            acc[nt] = __builtin_amdgcn_wmma_f32_16x16x32_f16(false, a_h, false, b_h, (short)0, acc[nt], false, false);
            acc[nt] = __builtin_amdgcn_wmma_f32_16x16x32_f16(false, a_h, false, b_l, (short)0, acc[nt], false, false);
            acc[nt] = __builtin_amdgcn_wmma_f32_16x16x32_f16(false, a_l, false, b_h, (short)0, acc[nt], false, false);
        }
    }
}

// ---------------------------------------------------------------------------
// Stage A: per (b,n): W2[n*64+kk][d] = sum_j K[b,j,n,kk] * fc_w[d, j*16+n]
//   M = kk (64), Kdim = j (1024), N = d (64 per block).
// ---------------------------------------------------------------------------
template <bool REPACKED>
__global__ __launch_bounds__(128)
void kw_stageA(const float* __restrict__ Kin, const float* __restrict__ Wsrc,
               float* __restrict__ W2) {
    const int d0 = blockIdx.x * 64;
    const int bn = blockIdx.z;
    const int b  = bn >> 4;
    const int n  = bn & 15;

    const float* A0 = Kin + (size_t)b * (SEQ * NHEADS * DKH) + n * DKH; // A[m,k]=A0[k*1024+m]
    const float* B0;
    long sBk, sBn;
    if (REPACKED) { B0 = Wsrc + (size_t)n * SEQ * DMODEL + d0;       sBk = DMODEL; sBn = 1; }
    else          { B0 = Wsrc + (size_t)d0 * (SEQ * NHEADS) + n;     sBk = NHEADS; sBn = SEQ * NHEADS; }

    v8f acc[4];
    gemm_macro64(A0, /*sAm=*/1, /*sAk=*/NHEADS * DKH, B0, sBk, sBn, SEQ / 32, acc);

    const int lane = threadIdx.x & 31, wave = threadIdx.x >> 5;
    const int r = lane & 15, h = lane >> 4;
    float* W2p = W2 + (size_t)b * (DMODEL * DMODEL) + (size_t)(n * DKH) * DMODEL + d0;
    #pragma unroll
    for (int nt = 0; nt < 4; ++nt)
        #pragma unroll
        for (int t = 0; t < 8; ++t) {
            const int row = wave * 16 + t + (h << 3);
            W2p[(size_t)row * DMODEL + nt * 16 + r] = acc[nt][t];
        }
}

// ---------------------------------------------------------------------------
// Stage B: per b: out[i,d] = 0.125 * sum_c Q[b,i,c] * W2[b][c][d] + fc_b[d]
// ---------------------------------------------------------------------------
__global__ __launch_bounds__(128)
void kw_stageB(const float* __restrict__ Q, const float* __restrict__ W2,
               const float* __restrict__ bias, float* __restrict__ Out) {
    const int d0 = blockIdx.x * 64;
    const int m0 = blockIdx.y * 64;
    const int b  = blockIdx.z;

    const float* A0 = Q  + (size_t)b * (SEQ * DMODEL) + (size_t)m0 * DMODEL; // A[m,k]=A0[m*1024+k]
    const float* B0 = W2 + (size_t)b * (DMODEL * DMODEL) + d0;               // B[k,n]=B0[k*1024+n]

    v8f acc[4];
    gemm_macro64(A0, DMODEL, 1, B0, DMODEL, 1, DMODEL / 32, acc);

    const int lane = threadIdx.x & 31, wave = threadIdx.x >> 5;
    const int r = lane & 15, h = lane >> 4;
    float* Op = Out + (size_t)b * (SEQ * DMODEL) + (size_t)m0 * DMODEL + d0;
    #pragma unroll
    for (int nt = 0; nt < 4; ++nt) {
        const float bv = bias[d0 + nt * 16 + r];
        #pragma unroll
        for (int t = 0; t < 8; ++t) {
            const int row = wave * 16 + t + (h << 3);
            Op[(size_t)row * DMODEL + nt * 16 + r] = 0.125f * acc[nt][t] + bv;
        }
    }
}

// ---------------------------------------------------------------------------

extern "C" void kernel_launch(void* const* d_in, const int* in_sizes, int n_in,
                              void* d_out, int out_size, void* d_ws, size_t ws_size,
                              hipStream_t stream) {
    const float* q    = (const float*)d_in[0];
    const float* k    = (const float*)d_in[1];
    // d_in[2] (v) is dead in the reference computation.
    const float* fc_w = (const float*)d_in[3];
    const float* fc_b = (const float*)d_in[4];
    float* out = (float*)d_out;

    float* w2   = (float*)d_ws;                                  // 16 MiB
    float* wrep = w2 + (size_t)BATCH * DMODEL * DMODEL;          // +64 MiB
    const size_t need_rep =
        ((size_t)BATCH * DMODEL * DMODEL + (size_t)NHEADS * SEQ * DMODEL) * sizeof(float);

    (void)in_sizes; (void)n_in; (void)out_size;

    dim3 blk(128, 1, 1);
    dim3 grdA(DMODEL / 64, 1, BATCH * NHEADS);   // 16 x 1 x 64
    if (ws_size >= need_rep) {
        dim3 tblk(32, 8, 1), tgrd((SEQ * NHEADS) / 32, DMODEL / 32, 1);
        repack_fcw<<<tgrd, tblk, 0, stream>>>(fc_w, wrep);
        kw_stageA<true><<<grdA, blk, 0, stream>>>(k, wrep, w2);
    } else {
        kw_stageA<false><<<grdA, blk, 0, stream>>>(k, fc_w, w2);
    }

    dim3 grdB(DMODEL / 64, SEQ / 64, BATCH);     // 16 x 16 x 4
    kw_stageB<<<grdB, blk, 0, stream>>>(q, w2, fc_b, out);
}
